// MultiHeadAttention_16320875725174
// MI455X (gfx1250) — compile-verified, matrix operand order checked by reading, not run on hardware
//
#include <hip/hip_runtime.h>
#include <hip/hip_bf16.h>

typedef __attribute__((ext_vector_type(16))) __bf16 v16bf;
typedef __attribute__((ext_vector_type(8)))  float  v8f;

static __device__ __forceinline__ v8f wmma_bf16(const v16bf& a, const v16bf& b, const v8f& c) {
    // D = A(16x32 bf16) * B(32x16 bf16) + C(16x16 f32)
    return __builtin_amdgcn_wmma_f32_16x16x32_bf16(false, a, false, b, (short)0, c, false, false);
}

// Async global->LDS 16-byte copy (GLOBAL_LOAD_ASYNC_TO_LDS_B128, ASYNCcnt).
static __device__ __forceinline__ void async_copy_b128(unsigned lds_addr, const void* gaddr) {
    asm volatile("global_load_async_to_lds_b128 %0, %1, off"
                 :: "v"(lds_addr), "v"((unsigned long long)(uintptr_t)gaddr)
                 : "memory");
}
// Async loads complete in order: waiting ASYNCcnt<=N with N in-flight prefetch
// ops guarantees all older (current-buffer) copies have landed.
template <int N>
static __device__ __forceinline__ void wait_asynccnt() {
    asm volatile("s_wait_asynccnt %0" :: "i"(N) : "memory");
}
static __device__ __forceinline__ unsigned lds_off(const void* p) {
    return (unsigned)(uintptr_t)p;   // generic LDS ptr: low 32 bits = LDS offset
}

// ---------------------------------------------------------------------------
// fp32 -> bf16 conversion pass (bandwidth-trivial; done once per tensor).
// ---------------------------------------------------------------------------
__global__ __launch_bounds__(256) void cvt_bf16_kernel(
    const float* __restrict__ in, __bf16* __restrict__ out, int n4)
{
    int i = blockIdx.x * blockDim.x + threadIdx.x;
    if (i < n4) {
        float4 f = ((const float4*)in)[i];
        int o = i * 4;
        out[o + 0] = (__bf16)f.x;
        out[o + 1] = (__bf16)f.y;
        out[o + 2] = (__bf16)f.z;
        out[o + 3] = (__bf16)f.w;
    }
}

// ---------------------------------------------------------------------------
// Tiled bf16 WMMA GEMM: C[M,N] = A[M,K] @ B[K,N] (+bias)
// Block: 256 threads (8 waves). Block tile 128x128, wave tile 64x32,
// K stepped by 32. Double-buffered LDS tiles filled with async global->LDS
// DMA overlapped against the WMMA phase of the previous tile.
// QKV_STORE: store bf16 into [b, h, t, d] layout; else fp32 row-major + bias.
// ---------------------------------------------------------------------------
template <bool QKV_STORE>
__global__ __launch_bounds__(256) void gemm128_kernel(
    const __bf16* __restrict__ A, const __bf16* __restrict__ B,
    const float* __restrict__ bias, void* __restrict__ Cv,
    int M, int K, int N)
{
    __shared__ __bf16 As[2][128][40];   // row stride 80 B (16B multiple)
    __shared__ __bf16 Bs[2][32][136];   // row stride 272 B (16B multiple)

    const int tid  = threadIdx.x;
    const int lane = tid & 31;
    const int wave = tid >> 5;
    const int half = lane >> 4;
    const int ln   = lane & 15;
    const int kb   = half * 8;

    const int blockN = blockIdx.x * 128;
    const int blockM = blockIdx.y * 128;
    const int waveM  = (wave >> 2) * 64;
    const int waveN  = (wave & 3) * 32;

    // per-thread chunk coordinates for the async stage (4 x b128 per thread)
    const int ar0 = tid >> 2,       ac0 = (tid & 3) * 8;          // A: 4 chunks/row
    const int ar1 = (tid + 256) >> 2, ac1 = ((tid + 256) & 3) * 8;
    const int br0 = tid >> 4,       bc0 = (tid & 15) * 8;         // B: 16 chunks/row
    const int br1 = (tid + 256) >> 4, bc1 = ((tid + 256) & 15) * 8;

    v8f acc[4][2];
    #pragma unroll
    for (int i = 0; i < 4; ++i)
        #pragma unroll
        for (int j = 0; j < 2; ++j)
            #pragma unroll
            for (int e = 0; e < 8; ++e) acc[i][j][e] = 0.0f;

    auto issue_tiles = [&](int k0, int buf) {
        async_copy_b128(lds_off(&As[buf][ar0][ac0]), A + (size_t)(blockM + ar0) * K + k0 + ac0);
        async_copy_b128(lds_off(&As[buf][ar1][ac1]), A + (size_t)(blockM + ar1) * K + k0 + ac1);
        async_copy_b128(lds_off(&Bs[buf][br0][bc0]), B + (size_t)(k0 + br0) * N + blockN + bc0);
        async_copy_b128(lds_off(&Bs[buf][br1][bc1]), B + (size_t)(k0 + br1) * N + blockN + bc1);
    };

    const int nk = K / 32;
    issue_tiles(0, 0);                       // prologue: fill buffer 0

    for (int t = 0; t < nk; ++t) {
        const int cur = t & 1;
        if (t + 1 < nk) {
            issue_tiles((t + 1) * 32, cur ^ 1);  // prefetch next tile (overlaps WMMA)
            wait_asynccnt<4>();                  // this wave's current-buffer copies done
        } else {
            wait_asynccnt<0>();
        }
        __syncthreads();                         // all waves' copies visible

        // ---- fragments per ISA 7.12.2 16-bit layouts ----
        v16bf aF[4], bF[2];
        #pragma unroll
        for (int mi = 0; mi < 4; ++mi) {
            int m = waveM + mi * 16 + ln;
            #pragma unroll
            for (int j = 0; j < 16; ++j) {
                int kk = (j < 8) ? (kb + j) : (kb + 8 + j);
                aF[mi][j] = As[cur][m][kk];
            }
        }
        #pragma unroll
        for (int ni = 0; ni < 2; ++ni) {
            int n = waveN + ni * 16 + ln;
            #pragma unroll
            for (int j = 0; j < 16; ++j) {
                int kk = (j < 8) ? (kb + j) : (kb + 8 + j);
                bF[ni][j] = Bs[cur][kk][n];
            }
        }
        #pragma unroll
        for (int mi = 0; mi < 4; ++mi)
            #pragma unroll
            for (int ni = 0; ni < 2; ++ni)
                acc[mi][ni] = wmma_bf16(aF[mi], bF[ni], acc[mi][ni]);

        __syncthreads();   // compute done before buffer 'cur' is refilled next iter
    }

    // ---- store: C/D layout: vgpr r -> row r + 8*half; col = ln ----
    #pragma unroll
    for (int mi = 0; mi < 4; ++mi) {
        #pragma unroll
        for (int ni = 0; ni < 2; ++ni) {
            #pragma unroll
            for (int r = 0; r < 8; ++r) {
                int m = blockM + waveM + mi * 16 + r + half * 8;
                int n = blockN + waveN + ni * 16 + ln;
                float v = acc[mi][ni][r];
                if (QKV_STORE) {
                    int b = m >> 11, t2 = m & 2047;
                    int h = n >> 6,  d = n & 63;
                    ((__bf16*)Cv)[(((size_t)b * 16 + h) * 2048 + t2) * 64 + d] = (__bf16)v;
                } else {
                    ((float*)Cv)[(size_t)m * N + n] = v + bias[n];
                }
            }
        }
    }
}

// ---------------------------------------------------------------------------
// Flash attention, causal, head_dim=64, n=2048.
// Block: 256 threads = 8 waves; each wave owns 16 query rows (block: 128).
// Double-buffered K/V tiles via async global->LDS DMA overlapped with the
// WMMA + softmax phase; online softmax with shfl_xor row reductions; P
// transposed to A-frag layout via per-wave LDS scratch.
// ---------------------------------------------------------------------------
__global__ __launch_bounds__(256) void attn_kernel(
    const __bf16* __restrict__ Q, const __bf16* __restrict__ Km,
    const __bf16* __restrict__ V, __bf16* __restrict__ ctx)
{
    __shared__ __bf16 Ks[2][32][72];    // row stride 144 B (16B multiple)
    __shared__ __bf16 Vs[2][32][72];
    __shared__ __bf16 Ps[8][16][40];    // per-wave P transpose scratch

    const int tid  = threadIdx.x;
    const int lane = tid & 31;
    const int wave = tid >> 5;
    const int half = lane >> 4;
    const int ln   = lane & 15;
    const int kb   = half * 8;

    const int bh = blockIdx.y;            // b*16+h
    const int b  = bh >> 4, h = bh & 15;
    const int qt = blockIdx.x;
    const int q0 = qt * 128 + wave * 16;

    const size_t base = (size_t)bh * 2048 * 64;

    // per-thread chunk coordinates for K/V staging (1 x b128 each per thread)
    const int cr = tid >> 3, cc = (tid & 7) * 8;   // 8 chunks per 128B row

    auto issue_kv = [&](int kt, int buf) {
        size_t g = base + (size_t)(kt * 32 + cr) * 64 + cc;
        async_copy_b128(lds_off(&Ks[buf][cr][cc]), Km + g);
        async_copy_b128(lds_off(&Vs[buf][cr][cc]), V + g);
    };

    // Q tile 16x64 -> two 16x32 A-fragments (resident all kernel)
    v16bf aQ[2];
    #pragma unroll
    for (int s = 0; s < 2; ++s) {
        int m = q0 + ln;
        #pragma unroll
        for (int j = 0; j < 16; ++j) {
            int d = s * 32 + ((j < 8) ? (kb + j) : (kb + 8 + j));
            aQ[s][j] = Q[base + (size_t)m * 64 + d];
        }
    }

    float mstat[8], lsum[8];
    #pragma unroll
    for (int r = 0; r < 8; ++r) { mstat[r] = -3.0e38f; lsum[r] = 0.0f; }
    v8f o[4];
    #pragma unroll
    for (int c = 0; c < 4; ++c)
        #pragma unroll
        for (int e = 0; e < 8; ++e) o[c][e] = 0.0f;

    const float sm_scale = 0.125f;   // 1/sqrt(64)
    const int ntiles = qt * 4 + 4;

    issue_kv(0, 0);                  // prologue: fill buffer 0

    for (int kt = 0; kt < ntiles; ++kt) {
        const int cur = kt & 1;
        if (kt + 1 < ntiles) {
            issue_kv(kt + 1, cur ^ 1);   // prefetch next K/V tile (overlaps compute)
            wait_asynccnt<2>();
        } else {
            wait_asynccnt<0>();
        }
        __syncthreads();

        if (kt * 32 <= q0 + 15) {    // skip fully-masked tiles (barriers stay uniform)
            // ---- S = Q @ K^T : 16 queries x 32 keys ----
            v8f s0, s1;
            #pragma unroll
            for (int e = 0; e < 8; ++e) { s0[e] = 0.0f; s1[e] = 0.0f; }
            #pragma unroll
            for (int s = 0; s < 2; ++s) {
                v16bf bK0, bK1;
                #pragma unroll
                for (int j = 0; j < 16; ++j) {
                    int d = s * 32 + ((j < 8) ? (kb + j) : (kb + 8 + j));
                    bK0[j] = Ks[cur][ln][d];
                    bK1[j] = Ks[cur][16 + ln][d];
                }
                s0 = wmma_bf16(aQ[s], bK0, s0);
                s1 = wmma_bf16(aQ[s], bK1, s1);
            }

            // ---- causal mask + online softmax ----
            float pr0[8], pr1[8], scl[8];
            #pragma unroll
            for (int r = 0; r < 8; ++r) {
                int qg  = q0 + r + half * 8;
                int kg0 = kt * 32 + ln;
                float v0 = (kg0      <= qg) ? s0[r] * sm_scale : -3.0e38f;
                float v1 = (kg0 + 16 <= qg) ? s1[r] * sm_scale : -3.0e38f;
                float mx = fmaxf(v0, v1);
                #pragma unroll
                for (int x = 1; x < 16; x <<= 1)
                    mx = fmaxf(mx, __shfl_xor(mx, x, 32));
                float mnew = fmaxf(mstat[r], mx);
                float sc = __expf(mstat[r] - mnew);
                float p0 = __expf(v0 - mnew);
                float p1 = __expf(v1 - mnew);
                float rs = p0 + p1;
                #pragma unroll
                for (int x = 1; x < 16; x <<= 1)
                    rs += __shfl_xor(rs, x, 32);
                lsum[r]  = lsum[r] * sc + rs;
                mstat[r] = mnew;
                scl[r] = sc; pr0[r] = p0; pr1[r] = p1;
            }

            #pragma unroll
            for (int c = 0; c < 4; ++c)
                #pragma unroll
                for (int r = 0; r < 8; ++r) o[c][r] *= scl[r];

            // ---- transpose P (C-layout -> A-frag layout) via LDS ----
            #pragma unroll
            for (int r = 0; r < 8; ++r) {
                Ps[wave][r + half * 8][ln]      = (__bf16)pr0[r];
                Ps[wave][r + half * 8][16 + ln] = (__bf16)pr1[r];
            }
            v16bf aP;
            #pragma unroll
            for (int j = 0; j < 16; ++j) {
                int kk = (j < 8) ? (kb + j) : (kb + 8 + j);
                aP[j] = Ps[wave][ln][kk];    // same-wave LDS RAW: in-order
            }
            // ---- ctx += P(16x32) @ V(32x64) ----
            #pragma unroll
            for (int c = 0; c < 4; ++c) {
                v16bf bV;
                #pragma unroll
                for (int j = 0; j < 16; ++j) {
                    int kk = (j < 8) ? (kb + j) : (kb + 8 + j);
                    bV[j] = Vs[cur][kk][c * 16 + ln];
                }
                o[c] = wmma_bf16(aP, bV, o[c]);
            }
        }
        __syncthreads();   // reads of buffer 'cur' done before it is refilled
    }

    // ---- normalize, store ctx bf16 as [b, t, h*64+d] for the output GEMM ----
    float inv[8];
    #pragma unroll
    for (int r = 0; r < 8; ++r) inv[r] = 1.0f / lsum[r];
    #pragma unroll
    for (int c = 0; c < 4; ++c) {
        #pragma unroll
        for (int r = 0; r < 8; ++r) {
            int t   = q0 + r + half * 8;
            int col = h * 64 + c * 16 + ln;
            ctx[((size_t)b * 2048 + t) * 1024 + col] = (__bf16)(o[c][r] * inv[r]);
        }
    }
}

// ---------------------------------------------------------------------------
extern "C" void kernel_launch(void* const* d_in, const int* in_sizes, int n_in,
                              void* d_out, int out_size, void* d_ws, size_t ws_size,
                              hipStream_t stream)
{
    const float* x  = (const float*)d_in[0];
    const float* Wq = (const float*)d_in[1];
    const float* Wk = (const float*)d_in[2];
    const float* Wv = (const float*)d_in[3];
    const float* Wo = (const float*)d_in[4];
    const float* bo = (const float*)d_in[5];

    const int M = 4 * 2048;            // 8192 tokens
    const int D = 1024;
    const size_t MD = (size_t)M * D;   // 8.39M elems
    const size_t DD = (size_t)D * D;   // 1.05M elems

    // Workspace (bf16): xb | Wq Wk Wv Wo | q k v ; ctx reuses xb region
    __bf16* xb  = (__bf16*)d_ws;
    __bf16* wqb = xb  + MD;
    __bf16* wkb = wqb + DD;
    __bf16* wvb = wkb + DD;
    __bf16* wob = wvb + DD;
    __bf16* q   = wob + DD;
    __bf16* k   = q + MD;
    __bf16* v   = k + MD;
    __bf16* ctx = xb;                  // safe: QKV GEMMs done before attention

    dim3 blk(256);

    // one-shot fp32 -> bf16 conversions
    cvt_bf16_kernel<<<(int)(MD / 4 / 256), blk, 0, stream>>>(x,  xb,  (int)(MD / 4));
    cvt_bf16_kernel<<<(int)(DD / 4 / 256), blk, 0, stream>>>(Wq, wqb, (int)(DD / 4));
    cvt_bf16_kernel<<<(int)(DD / 4 / 256), blk, 0, stream>>>(Wk, wkb, (int)(DD / 4));
    cvt_bf16_kernel<<<(int)(DD / 4 / 256), blk, 0, stream>>>(Wv, wvb, (int)(DD / 4));
    cvt_bf16_kernel<<<(int)(DD / 4 / 256), blk, 0, stream>>>(Wo, wob, (int)(DD / 4));

    dim3 gemmGrid(D / 128, M / 128);
    gemm128_kernel<true><<<gemmGrid, blk, 0, stream>>>(xb, wqb, nullptr, q, M, D, D);
    gemm128_kernel<true><<<gemmGrid, blk, 0, stream>>>(xb, wkb, nullptr, k, M, D, D);
    gemm128_kernel<true><<<gemmGrid, blk, 0, stream>>>(xb, wvb, nullptr, v, M, D, D);

    dim3 attnGrid(2048 / 128, 64);
    attn_kernel<<<attnGrid, blk, 0, stream>>>(q, k, v, ctx);

    gemm128_kernel<false><<<gemmGrid, blk, 0, stream>>>(ctx, wob, bo, (float*)d_out, M, D, D);
}
